// SmplLite_85564338471362
// MI455X (gfx1250) — matrix operand order repaired and checked by reading, not compiled, not run on hardware
//
#include <hip/hip_runtime.h>

typedef __attribute__((ext_vector_type(16))) __bf16 v16bf;
typedef __attribute__((ext_vector_type(8)))  float  v8f;

#define B_N    1024
#define V_N    6890
#define J_N    52
#define NB_N   16
#define NPOSE  459
#define KEXT   480          // 15 k-steps of 32 (476 used: 459 pose + 16 betas + 1 const)
#define KS1    15
#define N1     20670        // V*3
#define NT1    1292         // ceil(N1/16)
#define BT1    64           // B/16
#define KS2    2            // K=52 padded to 64
#define MT2    431          // ceil(V/16)

__constant__ int c_par[J_N] = {
  -1,0,0,0,1,2,3,4,5,6,7,8,9,9,9,12,13,14,16,17,18,19,20,22,23,20,25,26,20,
  28,29,20,31,32,20,34,35,21,37,38,21,40,41,21,43,44,21,46,47,21,49,50};

// 16-bit A-operand layout (16x32): lane L<16 holds M=L, halves K = 0..7,16..23;
// lane L+16 holds K = 8..15,24..31  (ISA 7.12.2)
__device__ __forceinline__ int koff_a(int lane, int i) {
  return ((i >> 3) << 4) + ((lane >> 4) << 3) + (i & 7);
}
// 16-bit B-operand layout (32x16): lane L<16 holds N=L, halves K = 0..7,16..23;
// lane L+16 holds K = 8..15,24..31  (same k order as A)
__device__ __forceinline__ int koff_b(int lane, int i) {
  return ((i >> 3) << 4) + ((lane >> 4) << 3) + (i & 7);
}

__device__ __forceinline__ void bf16_split(float v, __bf16* hi, __bf16* lo, size_t idx) {
  __bf16 h = (__bf16)v;
  hi[idx] = h;
  lo[idx] = (__bf16)(v - (float)h);
}

// ---------------- rot6d -> R, extended feature vector ----------------
__global__ __launch_bounds__(256)
void k_rot_feat(const float* __restrict__ body_pose, const float* __restrict__ global_orient,
                const float* __restrict__ betas, float* __restrict__ R, float* __restrict__ fext) {
  int t = blockIdx.x * 256 + threadIdx.x;
  if (t >= B_N * J_N) return;
  int b = t / J_N, j = t - b * J_N;
  const float* d6 = (j == 0) ? (global_orient + (size_t)b * 6)
                             : (body_pose + ((size_t)b * (J_N - 1) + (j - 1)) * 6);
  float a10 = d6[0], a11 = d6[1], a12 = d6[2];
  float a20 = d6[3], a21 = d6[4], a22 = d6[5];
  float r1 = 1.0f / sqrtf(a10*a10 + a11*a11 + a12*a12);
  float b10 = a10*r1, b11 = a11*r1, b12 = a12*r1;
  float d = b10*a20 + b11*a21 + b12*a22;
  float p0 = a20 - d*b10, p1 = a21 - d*b11, p2 = a22 - d*b12;
  float r2 = 1.0f / sqrtf(p0*p0 + p1*p1 + p2*p2);
  float b20 = p0*r2, b21 = p1*r2, b22 = p2*r2;
  float b30 = b11*b22 - b12*b21, b31 = b12*b20 - b10*b22, b32 = b10*b21 - b11*b20;
  float Rm[9] = {b10,b11,b12, b20,b21,b22, b30,b31,b32};
  float* Rout = R + ((size_t)b * J_N + j) * 9;
  #pragma unroll
  for (int i = 0; i < 9; i++) Rout[i] = Rm[i];
  if (j > 0) {
    float* f = fext + (size_t)b * KEXT + (size_t)(j - 1) * 9;
    #pragma unroll
    for (int i = 0; i < 9; i++) f[i] = Rm[i] - ((i == 0 || i == 4 || i == 8) ? 1.0f : 0.0f);
  } else {
    float* f = fext + (size_t)b * KEXT;
    for (int k = 0; k < NB_N; k++) f[NPOSE + k] = betas[(size_t)b * NB_N + k];
    f[NPOSE + NB_N] = 1.0f;                              // v_template row
    for (int k = NPOSE + NB_N + 1; k < KEXT; k++) f[k] = 0.0f;
  }
}

// ---------------- kinematic chain -> corrected 3x4 transforms ----------------
__global__ __launch_bounds__(256)
void k_chain(const float* __restrict__ betas, const float* __restrict__ Jt,
             const float* __restrict__ JS, const float* __restrict__ R,
             float* __restrict__ A_raw, float* __restrict__ A_cor) {
  int b = blockIdx.x * 256 + threadIdx.x;
  if (b >= B_N) return;
  const float* bet = betas + (size_t)b * NB_N;
  for (int j = 0; j < J_N; j++) {
    float jj[3];
    #pragma unroll
    for (int c = 0; c < 3; c++) {
      float s = Jt[j * 3 + c];
      const float* js = JS + ((size_t)j * 3 + c) * NB_N;
      for (int k = 0; k < NB_N; k++) s += bet[k] * js[k];
      jj[c] = s;
    }
    int p = c_par[j];
    float rel[3];
    if (p < 0) { rel[0] = jj[0]; rel[1] = jj[1]; rel[2] = jj[2]; }
    else {
      #pragma unroll
      for (int c = 0; c < 3; c++) {
        float s = Jt[p * 3 + c];
        const float* js = JS + ((size_t)p * 3 + c) * NB_N;
        for (int k = 0; k < NB_N; k++) s += bet[k] * js[k];
        rel[c] = jj[c] - s;
      }
    }
    const float* Rj = R + ((size_t)b * J_N + j) * 9;
    float M[12];
    #pragma unroll
    for (int r = 0; r < 3; r++) {
      M[r*4+0] = Rj[r*3+0]; M[r*4+1] = Rj[r*3+1]; M[r*4+2] = Rj[r*3+2]; M[r*4+3] = rel[r];
    }
    float A[12];
    if (p < 0) {
      #pragma unroll
      for (int i = 0; i < 12; i++) A[i] = M[i];
    } else {
      const float* P = A_raw + ((size_t)b * J_N + p) * 12;
      float Pr[12];
      #pragma unroll
      for (int i = 0; i < 12; i++) Pr[i] = P[i];
      #pragma unroll
      for (int r = 0; r < 3; r++)
        #pragma unroll
        for (int c = 0; c < 4; c++)
          A[r*4+c] = Pr[r*4+0]*M[0*4+c] + Pr[r*4+1]*M[1*4+c] + Pr[r*4+2]*M[2*4+c]
                   + (c == 3 ? Pr[r*4+3] : 0.0f);
    }
    float* Ar = A_raw + ((size_t)b * J_N + j) * 12;
    #pragma unroll
    for (int i = 0; i < 12; i++) Ar[i] = A[i];
    float* Ac = A_cor + ((size_t)b * J_N + j) * 12;
    #pragma unroll
    for (int r = 0; r < 3; r++) {
      float tc = A[r*4+0]*jj[0] + A[r*4+1]*jj[1] + A[r*4+2]*jj[2];
      Ac[r*4+0] = A[r*4+0]; Ac[r*4+1] = A[r*4+1]; Ac[r*4+2] = A[r*4+2]; Ac[r*4+3] = A[r*4+3] - tc;
    }
  }
}

// ---------------- operand converters: f32 -> bf16 hi/lo, WMMA-swizzled ----------------
__global__ __launch_bounds__(256)
void k_conv_pf(const float* __restrict__ fext, __bf16* __restrict__ hi, __bf16* __restrict__ lo) {
  int t = blockIdx.x * 256 + threadIdx.x;
  if (t >= BT1 * KS1 * 512) return;
  int i = t & 15, lane = (t >> 4) & 31, tile = t >> 9;
  int btile = tile / KS1, ks = tile - btile * KS1;
  int b = btile * 16 + (lane & 15);
  int k = ks * 32 + koff_a(lane, i);
  bf16_split(fext[(size_t)b * KEXT + k], hi, lo, (size_t)t);
}

__global__ __launch_bounds__(256)
void k_conv_pd(const float* __restrict__ posedirs, const float* __restrict__ shapedirs,
               const float* __restrict__ v_template, __bf16* __restrict__ hi, __bf16* __restrict__ lo) {
  long t = (long)blockIdx.x * 256 + threadIdx.x;
  if (t >= (long)KS1 * NT1 * 512) return;
  int i = (int)(t & 15), lane = (int)((t >> 4) & 31), tile = (int)(t >> 9);
  int ks = tile / NT1, ntile = tile - ks * NT1;
  int n = ntile * 16 + (lane & 15);
  int k = ks * 32 + koff_b(lane, i);
  float v = 0.0f;
  if (n < N1) {
    if (k < NPOSE)              v = posedirs[(size_t)k * N1 + n];
    else if (k < NPOSE + NB_N)  v = shapedirs[(size_t)n * NB_N + (k - NPOSE)];
    else if (k == NPOSE + NB_N) v = v_template[n];
  }
  bf16_split(v, hi, lo, (size_t)t);
}

__global__ __launch_bounds__(256)
void k_conv_w(const float* __restrict__ W, __bf16* __restrict__ hi, __bf16* __restrict__ lo) {
  int t = blockIdx.x * 256 + threadIdx.x;
  if (t >= MT2 * KS2 * 512) return;
  int i = t & 15, lane = (t >> 4) & 31, tile = t >> 9;
  int mtile = tile / KS2, ks = tile - mtile * KS2;
  int m = mtile * 16 + (lane & 15);
  int k = ks * 32 + koff_a(lane, i);
  float v = (m < V_N && k < J_N) ? W[(size_t)m * J_N + k] : 0.0f;
  bf16_split(v, hi, lo, (size_t)t);
}

__global__ __launch_bounds__(256)
void k_conv_amat(const float* __restrict__ A_cor, __bf16* __restrict__ hi, __bf16* __restrict__ lo) {
  int t = blockIdx.x * 256 + threadIdx.x;
  if (t >= B_N * KS2 * 512) return;
  int i = t & 15, lane = (t >> 4) & 31, tile = t >> 9;
  int b = tile / KS2, ks = tile - b * KS2;
  int n = lane & 15;
  int k = ks * 32 + koff_b(lane, i);
  float v = (k < J_N && n < 12) ? A_cor[((size_t)b * J_N + k) * 12 + n] : 0.0f;
  bf16_split(v, hi, lo, (size_t)t);
}

// ---------------- GEMM1: v_posed = fext @ [posedirs;shapedirs;v_template]  ----------------
__global__ __launch_bounds__(256)
void k_gemm1(const __bf16* __restrict__ pfh, const __bf16* __restrict__ pfl,
             const __bf16* __restrict__ pdh, const __bf16* __restrict__ pdl,
             float* __restrict__ vposed) {
  int wave = (blockIdx.x * 256 + threadIdx.x) >> 5;
  int lane = threadIdx.x & 31;
  int btile = wave / NT1, ntile = wave - btile * NT1;
  v8f c = {0.f,0.f,0.f,0.f,0.f,0.f,0.f,0.f};
  const __bf16* pAh = pfh + (size_t)btile * KS1 * 512 + lane * 16;
  const __bf16* pAl = pfl + (size_t)btile * KS1 * 512 + lane * 16;
  const __bf16* pBh = pdh + (size_t)ntile * 512 + lane * 16;
  const __bf16* pBl = pdl + (size_t)ntile * 512 + lane * 16;
  for (int ks = 0; ks < KS1; ks++) {
    v16bf ah = *(const v16bf*)(pAh + (size_t)ks * 512);
    v16bf al = *(const v16bf*)(pAl + (size_t)ks * 512);
    v16bf bh = *(const v16bf*)(pBh + (size_t)ks * NT1 * 512);
    v16bf bl = *(const v16bf*)(pBl + (size_t)ks * NT1 * 512);
    c = __builtin_amdgcn_wmma_f32_16x16x32_bf16(false, ah, false, bh, (short)0, c, false, false);
    c = __builtin_amdgcn_wmma_f32_16x16x32_bf16(false, ah, false, bl, (short)0, c, false, false);
    c = __builtin_amdgcn_wmma_f32_16x16x32_bf16(false, al, false, bh, (short)0, c, false, false);
  }
  int n = ntile * 16 + (lane & 15);
  int mbase = btile * 16 + ((lane >> 4) << 3);
  if (n < N1) {
    #pragma unroll
    for (int r = 0; r < 8; r++) vposed[(size_t)(mbase + r) * N1 + n] = c[r];
  }
}

// ---------------- GEMM2 + skinning epilogue (in-place on d_out) ----------------
__global__ __launch_bounds__(256)
void k_gemm2(const __bf16* __restrict__ wh, const __bf16* __restrict__ wl,
             const __bf16* __restrict__ abh, const __bf16* __restrict__ abl,
             const float* __restrict__ transl, float* __restrict__ out) {
  __shared__ float Tl[8][16][17];
  int wslot = threadIdx.x >> 5;
  int lane  = threadIdx.x & 31;
  int wave  = (blockIdx.x * 256 + threadIdx.x) >> 5;
  int b = wave / MT2, mtile = wave - b * MT2;
  v8f c = {0.f,0.f,0.f,0.f,0.f,0.f,0.f,0.f};
  #pragma unroll
  for (int ks = 0; ks < KS2; ks++) {
    v16bf ah = *(const v16bf*)(wh  + (size_t)(mtile * KS2 + ks) * 512 + lane * 16);
    v16bf al = *(const v16bf*)(wl  + (size_t)(mtile * KS2 + ks) * 512 + lane * 16);
    v16bf bh = *(const v16bf*)(abh + (size_t)(b * KS2 + ks) * 512 + lane * 16);
    v16bf bl = *(const v16bf*)(abl + (size_t)(b * KS2 + ks) * 512 + lane * 16);
    c = __builtin_amdgcn_wmma_f32_16x16x32_bf16(false, ah, false, bh, (short)0, c, false, false);
    c = __builtin_amdgcn_wmma_f32_16x16x32_bf16(false, ah, false, bl, (short)0, c, false, false);
    c = __builtin_amdgcn_wmma_f32_16x16x32_bf16(false, al, false, bh, (short)0, c, false, false);
  }
  int n  = lane & 15;
  int mb = (lane >> 4) << 3;
  #pragma unroll
  for (int r = 0; r < 8; r++) Tl[wslot][mb + r][n] = c[r];
  __syncthreads();
  if (lane < 16) {
    int v = mtile * 16 + lane;
    if (v < V_N) {
      float t[12];
      #pragma unroll
      for (int i = 0; i < 12; i++) t[i] = Tl[wslot][lane][i];
      size_t base = (size_t)b * N1 + (size_t)v * 3;
      float vx = out[base], vy = out[base + 1], vz = out[base + 2];
      float tx = transl[(size_t)b * 3 + 0];
      float ty = transl[(size_t)b * 3 + 1];
      float tz = transl[(size_t)b * 3 + 2];
      out[base + 0] = t[0]*vx + t[1]*vy + t[2]*vz  + t[3]  + tx;
      out[base + 1] = t[4]*vx + t[5]*vy + t[6]*vz  + t[7]  + ty;
      out[base + 2] = t[8]*vx + t[9]*vy + t[10]*vz + t[11] + tz;
    }
  }
}

extern "C" void kernel_launch(void* const* d_in, const int* in_sizes, int n_in,
                              void* d_out, int out_size, void* d_ws, size_t ws_size,
                              hipStream_t stream) {
  const float* body_pose     = (const float*)d_in[0];
  const float* betas         = (const float*)d_in[1];
  const float* global_orient = (const float*)d_in[2];
  const float* transl        = (const float*)d_in[3];
  const float* v_template    = (const float*)d_in[4];
  const float* shapedirs     = (const float*)d_in[5];
  const float* posedirs      = (const float*)d_in[6];
  const float* J_template    = (const float*)d_in[7];
  const float* J_shapedirs   = (const float*)d_in[8];
  const float* lbs_weights   = (const float*)d_in[9];
  float* out = (float*)d_out;
  char*  ws  = (char*)d_ws;

  size_t off = 0;
  auto take = [&](size_t bytes) -> char* {
    char* p = ws + off;
    off += (bytes + 255) & ~(size_t)255;
    return p;
  };
  float*  fext   = (float*)take((size_t)B_N * KEXT * 4);
  float*  Rws    = (float*)take((size_t)B_N * J_N * 9 * 4);
  float*  A_raw  = (float*)take((size_t)B_N * J_N * 12 * 4);
  float*  A_cor  = (float*)take((size_t)B_N * J_N * 12 * 4);
  __bf16* pf_hi  = (__bf16*)take((size_t)BT1 * KS1 * 512 * 2);
  __bf16* pf_lo  = (__bf16*)take((size_t)BT1 * KS1 * 512 * 2);
  __bf16* pd_hi  = (__bf16*)take((size_t)KS1 * NT1 * 512 * 2);
  __bf16* pd_lo  = (__bf16*)take((size_t)KS1 * NT1 * 512 * 2);
  __bf16* w_hi   = (__bf16*)take((size_t)MT2 * KS2 * 512 * 2);
  __bf16* w_lo   = (__bf16*)take((size_t)MT2 * KS2 * 512 * 2);
  __bf16* ab_hi  = (__bf16*)take((size_t)B_N * KS2 * 512 * 2);
  __bf16* ab_lo  = (__bf16*)take((size_t)B_N * KS2 * 512 * 2);

  k_rot_feat<<<(B_N * J_N + 255) / 256, 256, 0, stream>>>(body_pose, global_orient, betas, Rws, fext);
  k_chain<<<(B_N + 255) / 256, 256, 0, stream>>>(betas, J_template, J_shapedirs, Rws, A_raw, A_cor);
  k_conv_pf<<<(BT1 * KS1 * 512) / 256, 256, 0, stream>>>(fext, pf_hi, pf_lo);
  k_conv_pd<<<(KS1 * NT1 * 512) / 256, 256, 0, stream>>>(posedirs, shapedirs, v_template, pd_hi, pd_lo);
  k_conv_w<<<(MT2 * KS2 * 512 + 255) / 256, 256, 0, stream>>>(lbs_weights, w_hi, w_lo);
  k_conv_amat<<<(B_N * KS2 * 512) / 256, 256, 0, stream>>>(A_cor, ab_hi, ab_lo);
  // v_posed is written into d_out (saves 85 MB of ws), then skinning overwrites it in place.
  k_gemm1<<<(BT1 * NT1) / 8, 256, 0, stream>>>(pf_hi, pf_lo, pd_hi, pd_lo, out);
  k_gemm2<<<(B_N * MT2) / 8, 256, 0, stream>>>(w_hi, w_lo, ab_hi, ab_lo, transl, out);
}
